// RegressionLSTM_9698036154953
// MI455X (gfx1250) — matrix operand order repaired and checked by reading, not compiled, not run on hardware
//
#include <hip/hip_runtime.h>
#include <hip/hip_bf16.h>
#include <stdint.h>

// ---------------------------------------------------------------------------
// RegressionLSTM on MI455X (gfx1250)
//
// gates = concat(x_t, h_{t-1}) @ Wcat^T + b  in bf16 WMMA (f32 accumulate),
// f32 gate math, cell state c register-resident for the whole sequence.
// 8 layers pipelined across 8 x 4 workgroups (wavefront parallelism over the
// serial time recurrence), sync via agent-scope atomic step counters.
//
// CDNA5 paths: v_wmma_f32_16x16x32_bf16, TDM tensor_load_to_lds (staging the
// [64x512] bf16 activation tiles into LDS with hardware bank-padding),
// s_wait_tensorcnt, ds_load_b128 A-fragments, global_load_b128 + prefetch for
// pre-swizzled B-fragments.
//
// Assumes all 32 blocks resident and ws >= ~162MB.
// ---------------------------------------------------------------------------

#define LAYERS   8
#define HIDDEN   512
#define SENSORS  64
#define BATCH    64
#define SEQ      256
#define G4H      (4*HIDDEN)      // 2048 gate columns
#define NT       128             // 2048/16 N tiles
#define KTS      32              // K tile slots (16 x-region + 16 h-region)
#define WGPL     4               // workgroups per layer
#define THREADS  512             // 16 wave32

// LDS activation buffers: row = 512 bf16 (1024B) + 16B pad -> stride 520 elems
// (260 dwords, %64 = 4 -> 16 rows land on 16 distinct bank groups).
// The 16B pad is produced by the TDM descriptor (pad_interval=7, pad_amount=3).
#define AB_STRIDE 520
#define AB_BYTES  ((size_t)BATCH * AB_STRIDE * 2)   // 66,560 B per buffer
#define HB_OFF    ((uint32_t)AB_BYTES)              // hbuf LDS byte offset
#define GB_OFF    ((size_t)(2 * AB_BYTES))          // gate-exchange buffer
#define GB_STRIDE 130                                // padded f32 stride

typedef __bf16 bf16_t;
typedef __bf16       v16bf __attribute__((ext_vector_type(16)));
typedef float        v8f   __attribute__((ext_vector_type(8)));
typedef uint32_t     u32x4 __attribute__((ext_vector_type(4)));
typedef unsigned int ui32x4 __attribute__((ext_vector_type(4)));
typedef int          i32x4 __attribute__((ext_vector_type(4)));
typedef int          i32x8 __attribute__((ext_vector_type(8)));

// ---- workspace layout (bytes) ----------------------------------------------
#define WFRAG_DW   ((size_t)LAYERS*NT*KTS*256)                 // 8,388,608 dwords
#define OFF_WFRAG  ((size_t)0)
#define OFF_BIAS   (OFF_WFRAG + WFRAG_DW*4)                    // 33.5 MB
#define OFF_X0     (OFF_BIAS  + (size_t)LAYERS*G4H*4)
#define OFF_HSEQ   (OFF_X0    + (size_t)SEQ*BATCH*SENSORS*2)
#define OFF_DONE   (OFF_HSEQ  + (size_t)LAYERS*SEQ*BATCH*HIDDEN*2)  // +134 MB
// total ~162 MB

__device__ __forceinline__ v16bf mk16(u32x4 lo, u32x4 hi) {
  union { u32x4 u[2]; v16bf v; } t; t.u[0] = lo; t.u[1] = hi; return t.v;
}
__device__ __forceinline__ float sigm_f(float x) {
  return 1.0f / (1.0f + __expf(-x));
}
__device__ __forceinline__ float tanh_f(float x) {
  x = fminf(fmaxf(x, -15.0f), 15.0f);
  float e = __expf(2.0f * x);
  return (e - 1.0f) / (e + 1.0f);
}
__device__ __forceinline__ v8f wmma_bf16(v16bf a, v16bf b, v8f c) {
  return __builtin_amdgcn_wmma_f32_16x16x32_bf16(false, a, false, b,
                                                 (short)0, c, false, false);
}

// ---------------------------------------------------------------------------
// TDM: DMA a [64 x 512] bf16 row-major tile from global into LDS, inserting a
// 16B pad after every 1024B row (-> AB_STRIDE). D# per CDNA5 ISA ch.8:
//  group0: count=1 | lds_addr | global_addr(57b) | type=2
//  group1: data_size=2B, pad_enable, pad_interval=7 (256 dw), pad_amount=3 (4 dw)
//          tensor_dim0=512, tensor_dim1=64, tile_dim0=512, tile_dim1=64,
//          tensor_dim0_stride=512
//  groups 2/3 + trailing descriptor words: zero (plain 2D tensor)
// Issued by one wave; drained with s_wait_tensorcnt 0.
// This toolchain (clang-23 / therock-10.0) exposes the 6-arg builtin:
//   (uint32x4, int32x8, int32x4, int32x4, int32x8, i32 cpol)
// ---------------------------------------------------------------------------
__device__ __forceinline__ void tdm_load_h_tile(const bf16_t* src,
                                                uint32_t lds_byte_off) {
  uint64_t ga = (uint64_t)(uintptr_t)src;
  ui32x4 g0 = { 1u,                                   // count=1, user D#
                lds_byte_off,                         // LDS dest (bytes)
                (uint32_t)ga,                         // global addr [31:0]
                (uint32_t)((ga >> 32) & 0x01FFFFFFu)  // global addr [56:32]
                  | (2u << 30) };                     // type = 2 ("image")
  i32x8 g1;
  g1[0] = (1 << 16)        // data_size = 1 -> 2 bytes
        | (1 << 20)        // pad_enable
        | (7 << 22)        // pad_interval code 7 -> every 256 dwords (1024B)
        | (3 << 25);       // pad_amount  code 3 -> 4 dwords (16B)
  g1[1] = (int)(512u << 16);   // [31:16] tensor_dim0 lo16 (=512)
  g1[2] = (int)(64u  << 16);   // [15:0] tensor_dim0 hi16=0, [31:16] tensor_dim1=64
  g1[3] = (int)(512u << 16);   // [15:0] tensor_dim1 hi16=0, [31:16] tile_dim0=512
  g1[4] = 64;                  // [15:0] tile_dim1=64, [31:16] tile_dim2=0
  g1[5] = 512;                 // tensor_dim0_stride lo32
  g1[6] = 0;                   // stride0 hi16, stride1 lo16
  g1[7] = 0;                   // stride1 hi32
  i32x4 z4 = { 0, 0, 0, 0 };
  i32x8 z8 = { 0, 0, 0, 0, 0, 0, 0, 0 };
  __builtin_amdgcn_tensor_load_to_lds(g0, g1, z4, z4, z8, 0);
}

// ---------------------------------------------------------------------------
// Prep: build bf16 pre-swizzled WMMA B-fragments, combined bias, bf16 x (t-major)
// Dense 32x16 bf16 B lane layout (mirrors ISA sparse-B doc):
//   lanes 0-15 hold K=0..7,16..23, lanes 16-31 hold K=8..15,24..31 (VGPR v -> K=8h+2(v&3)+16(v>>2), +1).
// Fragment tile = 256 dwords, lane-linear: dword = tile*256 + lane*8 + v.
// ---------------------------------------------------------------------------
__global__ void lstm_prep(const float* __restrict__ x,
                          const float* __restrict__ Wih0,
                          const float* __restrict__ WihR,
                          const float* __restrict__ Whh,
                          const float* __restrict__ bih,
                          const float* __restrict__ bhh,
                          uint32_t* __restrict__ wfrag,
                          float*    __restrict__ bias,
                          bf16_t*   __restrict__ x0)
{
  const size_t N1 = WFRAG_DW;
  const size_t N2 = (size_t)LAYERS * G4H;
  const size_t N3 = (size_t)SEQ * BATCH * SENSORS;
  size_t d = (size_t)blockIdx.x * blockDim.x + threadIdx.x;
  if (d < N1) {
    uint32_t tile = (uint32_t)(d >> 8);
    uint32_t q    = (uint32_t)d & 255u;
    int lane = (int)(q >> 3), v = (int)(q & 7);
    int kt = tile & 31;
    int nt = (tile >> 5) & 127;
    int l  = (int)(tile >> 12);
    int side = lane >> 4, nl = lane & 15;
    int kin = side * 8 + (v & 3) * 2 + (v >> 2) * 16; // K of first packed element
    int n   = nt * 16 + nl;               // gate column 0..2047
    float f0 = 0.0f, f1 = 0.0f;
    if (kt < 16) {                        // x-region: B[k][n] = W_ih[n][k]
      int k = kt * 32 + kin;
      if (l == 0) {
        if (k < SENSORS) {
          f0 = Wih0[(size_t)n * SENSORS + k];
          f1 = Wih0[(size_t)n * SENSORS + k + 1];
        }
      } else {
        const float* W = WihR + (size_t)(l - 1) * G4H * HIDDEN;
        f0 = W[(size_t)n * HIDDEN + k];
        f1 = W[(size_t)n * HIDDEN + k + 1];
      }
    } else {                              // h-region: B[k][n] = W_hh[n][k]
      int k = (kt - 16) * 32 + kin;
      const float* W = Whh + (size_t)l * G4H * HIDDEN;
      f0 = W[(size_t)n * HIDDEN + k];
      f1 = W[(size_t)n * HIDDEN + k + 1];
    }
    union { bf16_t h[2]; uint32_t u; } p;
    p.h[0] = (bf16_t)f0; p.h[1] = (bf16_t)f1;
    wfrag[d] = p.u;
  } else if (d < N1 + N2) {
    size_t i = d - N1;
    bias[i] = bih[i] + bhh[i];
  } else if (d < N1 + N2 + N3) {
    size_t i = d - N1 - N2;               // (t, b, s) linear
    int t = (int)(i / (BATCH * SENSORS));
    int r = (int)(i % (BATCH * SENSORS));
    int b = r >> 6, s = r & 63;
    x0[i] = (bf16_t)x[((size_t)b * SEQ + t) * SENSORS + s];
  }
}

// ---------------------------------------------------------------------------
// Persistent pipelined LSTM. blockIdx: layer = bid/4, wg-in-layer g = bid%4.
// WG g owns h-columns [g*128, g*128+128). 16 waves:
//   waves 0..7  (gpair 0): gates (i,f), own cell state c, do the update
//   waves 8..15 (gpair 1): gates (g,o), hand over via LDS
// Each wave: 4 M-tiles x 2 gate-tiles = 8 f32 accum tiles; B-fragments unique
// per wave (no duplicated weight traffic inside a WG).
// ---------------------------------------------------------------------------
__global__ void __launch_bounds__(THREADS, 1)
lstm_pipeline(const uint32_t* __restrict__ wfrag,
              const float*    __restrict__ bias,
              const bf16_t*   __restrict__ x0,
              bf16_t*         __restrict__ hseq,
              int*            __restrict__ done)
{
  extern __shared__ char smem[];
  bf16_t* xbuf = (bf16_t*)smem;                 // [64][AB_STRIDE] x_t
  bf16_t* hbuf = (bf16_t*)(smem + AB_BYTES);    // [64][AB_STRIDE] h_{t-1}
  float*  gbuf = (float*)(smem + GB_OFF);       // [2][64][GB_STRIDE]

  const int bid  = blockIdx.x;
  const int l    = bid / WGPL;
  const int g    = bid % WGPL;
  const int tid  = threadIdx.x;
  const int w    = tid >> 5;
  const int lane = tid & 31;
  const int side = lane >> 4;
  const int nl   = lane & 15;
  const int gpair  = w >> 3;            // 0 -> (i,f), 1 -> (g,o)
  const int jt_loc = w & 7;
  const int jt     = g * 8 + jt_loc;    // global j tile 0..31
  const int kxt    = (l == 0) ? 2 : 16; // K-tiles in x-region

  const int gate0 = gpair * 2, gate1 = gpair * 2 + 1;
  const float b0 = bias[(size_t)l * G4H + gate0 * HIDDEN + jt * 16 + nl];
  const float b1 = bias[(size_t)l * G4H + gate1 * HIDDEN + jt * 16 + nl];

  const size_t btile0 = ((size_t)l * NT + gate0 * 32 + jt) * KTS; // frag tile base
  const size_t btile1 = ((size_t)l * NT + gate1 * 32 + jt) * KTS;

  // zero LDS once: provides h_{-1} = 0 and zero-padded layer0 x region
  for (int i = tid; i < 2 * BATCH * AB_STRIDE; i += THREADS) xbuf[i] = (bf16_t)0.0f;
  __syncthreads();

  v8f c0 = {}, c1 = {}, c2 = {}, c3 = {};   // cell state (gpair 0 only)

  bf16_t*       hl    = hseq + (size_t)l * SEQ * BATCH * HIDDEN;
  const bf16_t* hprev = hseq + (size_t)(l - 1) * SEQ * BATCH * HIDDEN;

  for (int t = 0; t < SEQ; ++t) {
    // ---- wait for producers (prev layer @ t, own layer @ t-1) -------------
    if (tid == 0) {
      if (l > 0)
        while (__hip_atomic_load(&done[(l - 1) * SEQ + t], __ATOMIC_ACQUIRE,
                                 __HIP_MEMORY_SCOPE_AGENT) < WGPL)
          __builtin_amdgcn_s_sleep(2);
      if (t > 0)
        while (__hip_atomic_load(&done[l * SEQ + (t - 1)], __ATOMIC_ACQUIRE,
                                 __HIP_MEMORY_SCOPE_AGENT) < WGPL)
          __builtin_amdgcn_s_sleep(2);
    }
    __syncthreads();

    // ---- stage x_t and h_{t-1} into LDS via the Tensor Data Mover ---------
    if (w == 0) {                         // one wave drives the TDM
      if (l > 0) tdm_load_h_tile(hprev + (size_t)t * BATCH * HIDDEN, 0u);
      if (t > 0) tdm_load_h_tile(hl + (size_t)(t - 1) * BATCH * HIDDEN, HB_OFF);
    }
    if (l == 0) {                         // tiny 8KB sensor tile: manual copy
      for (int i = tid; i < BATCH * SENSORS / 4; i += THREADS) {
        int row = (i * 4) / SENSORS, col = (i * 4) % SENSORS;
        *(unsigned long long*)&xbuf[row * AB_STRIDE + col] =
            *(const unsigned long long*)&x0[((size_t)t * BATCH + row) * SENSORS + col];
      }
    }
    if (w == 0) __builtin_amdgcn_s_wait_tensorcnt(0);
    __syncthreads();

    // ---- GEMM: 8 accum tiles per wave --------------------------------------
    v8f acc[4][2] = {};
    for (int kt = 0; kt < kxt; ++kt) {                   // x-region
      const int colbase = kt * 32 + side * 8;
      v16bf a[4];
      #pragma unroll
      for (int m = 0; m < 4; ++m) {
        const u32x4* ap = (const u32x4*)&xbuf[(m * 16 + nl) * AB_STRIDE + colbase];
        a[m] = mk16(ap[0], ap[2]);
      }
      const u32x4* bp0 = (const u32x4*)&wfrag[(btile0 + kt) * 256 + lane * 8];
      const u32x4* bp1 = (const u32x4*)&wfrag[(btile1 + kt) * 256 + lane * 8];
      __builtin_prefetch(bp0 + 64, 0, 1);
      __builtin_prefetch(bp1 + 64, 0, 1);
      v16bf bb0 = mk16(bp0[0], bp0[1]);
      v16bf bb1 = mk16(bp1[0], bp1[1]);
      #pragma unroll
      for (int m = 0; m < 4; ++m) {
        acc[m][0] = wmma_bf16(a[m], bb0, acc[m][0]);
        acc[m][1] = wmma_bf16(a[m], bb1, acc[m][1]);
      }
    }
    #pragma unroll 2
    for (int kt = 0; kt < 16; ++kt) {                    // h-region
      const int colbase = kt * 32 + side * 8;
      v16bf a[4];
      #pragma unroll
      for (int m = 0; m < 4; ++m) {
        const u32x4* ap = (const u32x4*)&hbuf[(m * 16 + nl) * AB_STRIDE + colbase];
        a[m] = mk16(ap[0], ap[2]);
      }
      const u32x4* bp0 = (const u32x4*)&wfrag[(btile0 + 16 + kt) * 256 + lane * 8];
      const u32x4* bp1 = (const u32x4*)&wfrag[(btile1 + 16 + kt) * 256 + lane * 8];
      __builtin_prefetch(bp0 + 64, 0, 1);
      __builtin_prefetch(bp1 + 64, 0, 1);
      v16bf bb0 = mk16(bp0[0], bp0[1]);
      v16bf bb1 = mk16(bp1[0], bp1[1]);
      #pragma unroll
      for (int m = 0; m < 4; ++m) {
        acc[m][0] = wmma_bf16(a[m], bb0, acc[m][0]);
        acc[m][1] = wmma_bf16(a[m], bb1, acc[m][1]);
      }
    }

    // ---- gate nonlinearity + cell update -----------------------------------
    const int cl = jt_loc * 16 + nl;     // local column 0..127
    if (gpair == 1) {                    // g = tanh, o = sigmoid -> LDS
      #pragma unroll
      for (int m = 0; m < 4; ++m) {
        #pragma unroll
        for (int r = 0; r < 8; ++r) {
          int b = m * 16 + r + side * 8; // C/D layout: lanes16-31 hold M=8+r
          gbuf[(size_t)b * GB_STRIDE + cl]        = tanh_f(acc[m][0][r] + b0);
          gbuf[(size_t)(64 + b) * GB_STRIDE + cl] = sigm_f(acc[m][1][r] + b1);
        }
      }
    }
    __syncthreads();
    if (gpair == 0) {                    // i,f in regs; c = f*c + i*g; h = o*tanh(c)
      bf16_t* hdst = hl + (size_t)t * BATCH * HIDDEN;
      #pragma unroll
      for (int m = 0; m < 4; ++m) {
        v8f* cc = (m == 0) ? &c0 : (m == 1) ? &c1 : (m == 2) ? &c2 : &c3;
        #pragma unroll
        for (int r = 0; r < 8; ++r) {
          int b = m * 16 + r + side * 8;
          float iv = sigm_f(acc[m][0][r] + b0);
          float fv = sigm_f(acc[m][1][r] + b1);
          float gv = gbuf[(size_t)b * GB_STRIDE + cl];
          float ov = gbuf[(size_t)(64 + b) * GB_STRIDE + cl];
          float cn = fv * (*cc)[r] + iv * gv;
          (*cc)[r] = cn;
          hdst[(size_t)b * HIDDEN + g * 128 + cl] = (bf16_t)(ov * tanh_f(cn));
        }
      }
    }
    __threadfence();
    __syncthreads();
    if (tid == 0)
      __hip_atomic_fetch_add(&done[l * SEQ + t], 1, __ATOMIC_RELEASE,
                             __HIP_MEMORY_SCOPE_AGENT);
  }
}

// ---------------------------------------------------------------------------
// out[b][o] = h7[T-1][b] . W_lin[o] + b_lin[o]
// ---------------------------------------------------------------------------
__global__ void final_linear(const bf16_t* __restrict__ hseq,
                             const float*  __restrict__ Wlin,
                             const float*  __restrict__ blin,
                             float*        __restrict__ out)
{
  int tid = threadIdx.x;
  if (tid >= BATCH * 2) return;
  int b = tid >> 1, o = tid & 1;
  const bf16_t* h = hseq + ((size_t)7 * SEQ + (SEQ - 1)) * BATCH * HIDDEN
                         + (size_t)b * HIDDEN;
  float s = blin[o];
  for (int j = 0; j < HIDDEN; ++j) s += (float)h[j] * Wlin[o * HIDDEN + j];
  out[b * 2 + o] = s;
}

// ---------------------------------------------------------------------------
extern "C" void kernel_launch(void* const* d_in, const int* in_sizes, int n_in,
                              void* d_out, int out_size, void* d_ws, size_t ws_size,
                              hipStream_t stream) {
  (void)in_sizes; (void)n_in; (void)out_size; (void)ws_size;
  const float* x    = (const float*)d_in[0];
  const float* Wih0 = (const float*)d_in[1];
  const float* WihR = (const float*)d_in[2];
  const float* Whh  = (const float*)d_in[3];
  const float* bih  = (const float*)d_in[4];
  const float* bhh  = (const float*)d_in[5];
  const float* Wlin = (const float*)d_in[6];
  const float* blin = (const float*)d_in[7];
  float* out = (float*)d_out;

  char* ws = (char*)d_ws;
  uint32_t* wfrag = (uint32_t*)(ws + OFF_WFRAG);
  float*    bias  = (float*)(ws + OFF_BIAS);
  bf16_t*   x0    = (bf16_t*)(ws + OFF_X0);
  bf16_t*   hseq  = (bf16_t*)(ws + OFF_HSEQ);
  int*      done  = (int*)(ws + OFF_DONE);

  // reset pipeline flags every launch (graph-capture safe, stream-ordered)
  (void)hipMemsetAsync(done, 0, (size_t)LAYERS * SEQ * sizeof(int), stream);

  size_t total = WFRAG_DW + (size_t)LAYERS * G4H + (size_t)SEQ * BATCH * SENSORS;
  int blocks = (int)((total + 255) / 256);
  lstm_prep<<<blocks, 256, 0, stream>>>(x, Wih0, WihR, Whh, bih, bhh,
                                        wfrag, bias, x0);

  size_t shmem = 2 * AB_BYTES + (size_t)2 * 64 * GB_STRIDE * 4;  // ~195KB
  (void)hipFuncSetAttribute(reinterpret_cast<const void*>(lstm_pipeline),
                            hipFuncAttributeMaxDynamicSharedMemorySize,
                            (int)shmem);
  lstm_pipeline<<<LAYERS * WGPL, THREADS, shmem, stream>>>(wfrag, bias, x0,
                                                           hseq, done);

  final_linear<<<1, 128, 0, stream>>>(hseq, Wlin, blin, out);
}
